// LSM_47236050321597
// MI455X (gfx1250) — matrix operand order repaired and checked by reading, not compiled, not run on hardware
//
#include <hip/hip_runtime.h>
#include <hip/hip_bf16.h>

typedef __attribute__((ext_vector_type(16))) __bf16 v16bf;
typedef __attribute__((ext_vector_type(8)))  __bf16 v8bf;
typedef __attribute__((ext_vector_type(8)))  float  v8f;

#define CDIM 128
#define HDIM 128
#define ROWS_PER_BLOCK 64       // 64-row slab per block; 4 x 16-row M-tiles
#define H1_STRIDE 136           // ushorts; 272B rows: 16B-aligned, bank-staggered

// ---------------------------------------------------------------------------
// Pack a 128x128 f32 weight (K x N, row major) into bf16 WMMA B-fragment
// layout: frag idx = (coltile*4 + kt)*32 + lane; element e has
// k = kt*32 + (lane>>4)*8 + e + (e>=8 ? 8 : 0), col = coltile*16 + (lane&15).
// The GEMM then fetches each B frag as one contiguous 32B load per lane.
// ---------------------------------------------------------------------------
__global__ __launch_bounds__(256) void pack_w_kernel(
    const float* __restrict__ W, unsigned short* __restrict__ Wp)
{
    const int idx = blockIdx.x * blockDim.x + threadIdx.x;   // 0..1023
    if (idx >= 8 * 4 * 32) return;
    const int lane = idx & 31;
    const int kt   = (idx >> 5) & 3;
    const int w    = idx >> 7;
    const int col  = w * 16 + (lane & 15);
    const int kb   = kt * 32 + (lane >> 4) * 8;
    #pragma unroll
    for (int e = 0; e < 16; ++e) {
        const __bf16 bv = (__bf16)W[(size_t)(kb + e + (e >= 8 ? 8 : 0)) * HDIM + col];
        Wp[(size_t)idx * 16 + e] = __builtin_bit_cast(unsigned short, bv);
    }
}

// ---------------------------------------------------------------------------
// Fused kernel: for a 64-row slab,
//   h1 = relu(x @ W1 + b1)          (bf16, LDS only)
//   h2 = relu(h1 @ W2 + b2)         (f32, LDS only)
//   s_q[n] = h2[n].We[0:128],  s_k[n] = h2[n].We[128:256]
// Neither intermediate touches HBM. Wave w owns column tile w; both weight
// matrices stay resident as packed bf16 fragments (one 32B load per frag).
// ---------------------------------------------------------------------------
__global__ __launch_bounds__(256) void fused_mlp_proj_kernel(
    const float* __restrict__ X,             // N x 128 f32
    const unsigned short* __restrict__ Wp1,  // packed bf16 B frags of W1
    const unsigned short* __restrict__ Wp2,  // packed bf16 B frags of W2
    const float* __restrict__ b1,
    const float* __restrict__ b2,
    const float* __restrict__ We,            // 256 = Wq | Wk
    float* __restrict__ sq,
    float* __restrict__ sk,
    int nrows)
{
    __shared__ unsigned short h1l[ROWS_PER_BLOCK][H1_STRIDE];   // ~17 KB
    __shared__ float          h2l[ROWS_PER_BLOCK][HDIM + 1];    // ~33 KB

    const int wave = threadIdx.x >> 5;
    const int lane = threadIdx.x & 31;
    const int half = lane >> 4;
    const int l15  = lane & 15;
    const int col  = wave * 16 + l15;
    const int row0 = blockIdx.x * ROWS_PER_BLOCK;

    // Resident B fragments for this wave's column tile, both layers.
    v16bf bf1[4], bf2[4];
    #pragma unroll
    for (int kt = 0; kt < 4; ++kt) {
        const size_t fo = (size_t)(((wave * 4 + kt) * 32 + lane)) * 16;
        bf1[kt] = *(const v16bf*)(Wp1 + fo);
        bf2[kt] = *(const v16bf*)(Wp2 + fo);
    }
    const float bias1 = b1[col];
    const float bias2 = b2[col];

    // ---- Phase 1: x @ W1 -> relu -> bf16 slab in LDS -------------------
    for (int mt = 0; mt < 4; ++mt) {
        int row = row0 + mt * 16 + l15;
        if (row > nrows - 1) row = nrows - 1;   // uniform clamp, no divergence
        const float* xrow = X + (size_t)row * CDIM;

        v8f acc = {};
        #pragma unroll
        for (int kt = 0; kt < 4; ++kt) {
            const int ka = kt * 32 + half * 8;
            const float4 f0 = *(const float4*)(xrow + ka);
            const float4 f1 = *(const float4*)(xrow + ka + 4);
            const float4 f2 = *(const float4*)(xrow + ka + 16);
            const float4 f3 = *(const float4*)(xrow + ka + 20);
            v16bf a;
            a[0]  = (__bf16)f0.x; a[1]  = (__bf16)f0.y; a[2]  = (__bf16)f0.z; a[3]  = (__bf16)f0.w;
            a[4]  = (__bf16)f1.x; a[5]  = (__bf16)f1.y; a[6]  = (__bf16)f1.z; a[7]  = (__bf16)f1.w;
            a[8]  = (__bf16)f2.x; a[9]  = (__bf16)f2.y; a[10] = (__bf16)f2.z; a[11] = (__bf16)f2.w;
            a[12] = (__bf16)f3.x; a[13] = (__bf16)f3.y; a[14] = (__bf16)f3.z; a[15] = (__bf16)f3.w;
            acc = __builtin_amdgcn_wmma_f32_16x16x32_bf16(
                false, a, false, bf1[kt], (short)0, acc, false, false);
        }

        #pragma unroll
        for (int r = 0; r < 8; ++r) {
            const float v = fmaxf(acc[r] + bias1, 0.0f);
            const __bf16 bv = (__bf16)v;
            h1l[mt * 16 + r + half * 8][col] = __builtin_bit_cast(unsigned short, bv);
        }
    }
    __syncthreads();

    // ---- Phase 2: h1 @ W2 -> relu -> f32 slab in LDS -------------------
    for (int mt = 0; mt < 4; ++mt) {
        const unsigned short* hrow = &h1l[mt * 16 + l15][0];

        v8f acc = {};
        #pragma unroll
        for (int kt = 0; kt < 4; ++kt) {
            const int ka = kt * 32 + half * 8;
            const v8bf lo = *(const v8bf*)(hrow + ka);
            const v8bf hi = *(const v8bf*)(hrow + ka + 16);
            const v16bf a = __builtin_shufflevector(lo, hi,
                0, 1, 2, 3, 4, 5, 6, 7, 8, 9, 10, 11, 12, 13, 14, 15);
            acc = __builtin_amdgcn_wmma_f32_16x16x32_bf16(
                false, a, false, bf2[kt], (short)0, acc, false, false);
        }

        #pragma unroll
        for (int r = 0; r < 8; ++r) {
            h2l[mt * 16 + r + half * 8][col] = fmaxf(acc[r] + bias2, 0.0f);
        }
    }
    __syncthreads();

    // ---- Fused projection: one thread per slab row ---------------------
    if (threadIdx.x < ROWS_PER_BLOCK) {
        const int r    = threadIdx.x;
        const int node = row0 + r;
        if (node < nrows) {
            float aq = 0.0f, ak = 0.0f;
            #pragma unroll 8
            for (int c = 0; c < HDIM; ++c) {
                const float hv = h2l[r][c];
                aq += hv * We[c];
                ak += hv * We[HDIM + c];
            }
            sq[node] = aq;
            sk[node] = ak;
        }
    }
}

// ---------------------------------------------------------------------------
// Streaming edge scoring + index passthrough.
//   out[0:E)   = sq[pos_src] + sk[pos_dst] + be
//   out[E:2E)  = sq[neg_src] + sk[neg_dst] + be
//   out[2E:4E) = float(edge_index_neg)
// sq/sk (400 KB) stay L2-resident -> pure streaming bandwidth.
// ---------------------------------------------------------------------------
__global__ __launch_bounds__(256) void edge_score_kernel(
    const int* __restrict__ ei_pos,
    const int* __restrict__ ei_neg,
    const float* __restrict__ sq,
    const float* __restrict__ sk,
    const float* __restrict__ be,
    float* __restrict__ out,
    int E)
{
    const int tid = blockIdx.x * blockDim.x + threadIdx.x;
    if (tid < E) {
        const float b = be[0];
        out[tid]             = sq[ei_pos[tid]] + sk[ei_pos[E + tid]] + b;
        out[(size_t)E + tid] = sq[ei_neg[tid]] + sk[ei_neg[E + tid]] + b;
    }
    if (tid < 2 * E) {
        out[2 * (size_t)E + tid] = (float)ei_neg[tid];
    }
}

extern "C" void kernel_launch(void* const* d_in, const int* in_sizes, int n_in,
                              void* d_out, int out_size, void* d_ws, size_t ws_size,
                              hipStream_t stream)
{
    const float* x   = (const float*)d_in[0];
    const int*   eip = (const int*)d_in[1];
    const int*   ein = (const int*)d_in[2];
    // d_in[3] = batch (unused by the reference outputs)
    const float* W1  = (const float*)d_in[4];
    const float* b1  = (const float*)d_in[5];
    const float* W2  = (const float*)d_in[6];
    const float* b2  = (const float*)d_in[7];
    const float* We  = (const float*)d_in[8];
    const float* be  = (const float*)d_in[9];

    const int N = in_sizes[0] / CDIM;   // 50000
    const int E = in_sizes[1] / 2;      // 1600000

    // Workspace: sq (N f32) | sk (N f32) | Wp1 | Wp2 (32KB each)
    size_t off = 0;
    float* sq = (float*)((char*)d_ws + off);
    off += ((size_t)N * sizeof(float) + 255) & ~(size_t)255;
    float* sk = (float*)((char*)d_ws + off);
    off += ((size_t)N * sizeof(float) + 255) & ~(size_t)255;
    unsigned short* Wp1 = (unsigned short*)((char*)d_ws + off);
    off += 8 * 4 * 32 * 16 * sizeof(unsigned short);
    unsigned short* Wp2 = (unsigned short*)((char*)d_ws + off);

    pack_w_kernel<<<4, 256, 0, stream>>>(W1, Wp1);
    pack_w_kernel<<<4, 256, 0, stream>>>(W2, Wp2);

    const int nblocks = (N + ROWS_PER_BLOCK - 1) / ROWS_PER_BLOCK;
    fused_mlp_proj_kernel<<<nblocks, 256, 0, stream>>>(x, Wp1, Wp2, b1, b2, We,
                                                       sq, sk, N);

    const int threads = 256;
    const int blocks  = (2 * E + threads - 1) / threads;
    edge_score_kernel<<<blocks, threads, 0, stream>>>(eip, ein, sq, sk, be,
                                                      (float*)d_out, E);
}
